// GAT_14147622273466
// MI455X (gfx1250) — matrix operand order repaired and checked by reading, not compiled, not run on hardware
//
#include <hip/hip_runtime.h>
#include <hip/hip_bf16.h>

// Problem constants (match reference)
#define T_DIM      64
#define N_DIM      4096
#define K_NEIGH    5
#define IN_DIM     256
#define OUT_DIM    256
#define M_TOTAL    (T_DIM * N_DIM)          // 262144 rows
#define ROWS_PB    32                        // rows per block (2 x 16-row WMMA tiles)
#define NUM_BLOCKS (M_TOTAL / ROWS_PB)       // 8192

typedef __bf16 bf16_t;
typedef __attribute__((ext_vector_type(16))) __bf16 v16bf;
typedef __attribute__((ext_vector_type(8)))  __bf16 v8bf;
typedef __attribute__((ext_vector_type(4)))  __bf16 v4bf;
typedef __attribute__((ext_vector_type(8)))  float  v8f;

union V16 { v16bf v; struct { v8bf lo, hi; } h; };

// ---------------------------------------------------------------------------
// Kernel 1: pack Wl / Wr (f32, [OUT,IN] row-major) into bf16 in workspace.
// Layout in ws: [ Wl_bf16 (65536) | Wr_bf16 (65536) ]  -> 256 KB, L2-resident.
// ---------------------------------------------------------------------------
__global__ __launch_bounds__(256) void pack_weights_bf16(
    const float* __restrict__ wl, const float* __restrict__ wr,
    bf16_t* __restrict__ wsb) {
  int t = blockIdx.x * 256 + threadIdx.x;               // grid = 512 -> t in [0,131072)
  float v = (t < OUT_DIM * IN_DIM) ? wl[t] : wr[t - OUT_DIM * IN_DIM];
  wsb[t] = (bf16_t)v;
}

// ---------------------------------------------------------------------------
// Kernel 2: fused  aggr = sum_k w[k]*neigh[:,k,:]  then
//           out   = x @ Wl^T + aggr @ Wr^T   via v_wmma_f32_16x16x32_bf16
// Block = 256 threads (8 waves), owns ROWS_PB=32 rows x 256 out cols.
// Wave w owns out cols [32w, 32w+32): 2x2 grid of 16x16 f32 accumulators.
// ---------------------------------------------------------------------------
__global__ __launch_bounds__(256) void gat_fused_wmma(
    const float* __restrict__ x,        // [M, IN_DIM]
    const float* __restrict__ neigh,    // [M, K, IN_DIM]
    const float* __restrict__ w_aggr,   // [K]
    const bf16_t* __restrict__ wl_b,    // [OUT, IN] bf16
    const bf16_t* __restrict__ wr_b,    // [OUT, IN] bf16
    float* __restrict__ out) {          // [M, OUT_DIM]
  __shared__ bf16_t ldsX[ROWS_PB * IN_DIM];   // 16 KB
  __shared__ bf16_t ldsA[ROWS_PB * IN_DIM];   // 16 KB

  const int tid     = threadIdx.x;
  const int rowbase = blockIdx.x * ROWS_PB;

  float wk[K_NEIGH];
#pragma unroll
  for (int k = 0; k < K_NEIGH; ++k) wk[k] = w_aggr[k];

  // ---------------- Phase 1: stream + reduce + cvt to bf16 in LDS ----------
  const size_t xbase = (size_t)rowbase * IN_DIM;
  const size_t nbase = (size_t)rowbase * (K_NEIGH * IN_DIM);
#pragma unroll
  for (int i = 0; i < (ROWS_PB * IN_DIM) / (256 * 4); ++i) {   // 8 iters
    const int e   = i * (256 * 4) + tid * 4;                   // tile element
    const int row = e >> 8;                                    // /IN_DIM
    const int d   = e & (IN_DIM - 1);

    const float4 xv = *(const float4*)(x + xbase + e);

    float4 acc = make_float4(0.f, 0.f, 0.f, 0.f);
    const float* np = neigh + nbase + (size_t)row * (K_NEIGH * IN_DIM) + d;
#pragma unroll
    for (int k = 0; k < K_NEIGH; ++k) {
      const float4 nv = *(const float4*)(np + k * IN_DIM);
      acc.x += wk[k] * nv.x;  acc.y += wk[k] * nv.y;
      acc.z += wk[k] * nv.z;  acc.w += wk[k] * nv.w;
    }

    v4bf xb, ab;
    xb[0] = (bf16_t)xv.x;  xb[1] = (bf16_t)xv.y;
    xb[2] = (bf16_t)xv.z;  xb[3] = (bf16_t)xv.w;
    ab[0] = (bf16_t)acc.x; ab[1] = (bf16_t)acc.y;
    ab[2] = (bf16_t)acc.z; ab[3] = (bf16_t)acc.w;
    *(v4bf*)(&ldsX[e]) = xb;    // ds_store_b64
    *(v4bf*)(&ldsA[e]) = ab;
  }

  __syncthreads();

  // ---------------- Phase 2: WMMA ------------------------------------------
  const int wave = tid >> 5;
  const int lane = tid & 31;
  const int l    = lane & 15;       // row-in-tile / col-in-tile selector
  const int hs   = lane >> 4;       // which half of the K range this lane holds
  const int colb = wave * 32;       // this wave's first output column

  v8f c00 = {}; v8f c01 = {}; v8f c10 = {}; v8f c11 = {};

#pragma unroll
  for (int kt = 0; kt < IN_DIM / 32; ++kt) {                 // 8 k-steps of 32
    // A fragments (16-bit A 16x32 layout, ISA 7.12.2):
    //   lanes 0-15 : M=l, K = {0..7, 16..23};  lanes 16-31: M=l, K = {8..15, 24..31}
    const int dA = kt * 32 + hs * 8;
    V16 ax0, ax1, aa0, aa1;
    {
      const bf16_t* p = &ldsX[(0 * 16 + l) * IN_DIM + dA];
      ax0.h.lo = *(const v8bf*)p;        ax0.h.hi = *(const v8bf*)(p + 16);
      p = &ldsX[(1 * 16 + l) * IN_DIM + dA];
      ax1.h.lo = *(const v8bf*)p;        ax1.h.hi = *(const v8bf*)(p + 16);
      p = &ldsA[(0 * 16 + l) * IN_DIM + dA];
      aa0.h.lo = *(const v8bf*)p;        aa0.h.hi = *(const v8bf*)(p + 16);
      p = &ldsA[(1 * 16 + l) * IN_DIM + dA];
      aa1.h.lo = *(const v8bf*)p;        aa1.h.hi = *(const v8bf*)(p + 16);
    }
    // B fragments (32x16 K-major): lanes 0-15: N=l, K={0..7,16..23}; lanes 16-31: N=l, K={8..15,24..31}
    // B[k][n] = W[o=n][d=k]  (out = act @ W^T), W rows are contiguous in d.
    const int dB = kt * 32 + hs * 8;
    V16 bl0, bl1, br0, br1;
    {
      const bf16_t* q = wl_b + (size_t)(colb + l) * IN_DIM + dB;
      bl0.h.lo = *(const v8bf*)q;        bl0.h.hi = *(const v8bf*)(q + 16);
      q = wl_b + (size_t)(colb + 16 + l) * IN_DIM + dB;
      bl1.h.lo = *(const v8bf*)q;        bl1.h.hi = *(const v8bf*)(q + 16);
      q = wr_b + (size_t)(colb + l) * IN_DIM + dB;
      br0.h.lo = *(const v8bf*)q;        br0.h.hi = *(const v8bf*)(q + 16);
      q = wr_b + (size_t)(colb + 16 + l) * IN_DIM + dB;
      br1.h.lo = *(const v8bf*)q;        br1.h.hi = *(const v8bf*)(q + 16);
    }

    // out = x@Wl^T + aggr@Wr^T : both products accumulate into the same C.
    c00 = __builtin_amdgcn_wmma_f32_16x16x32_bf16(false, ax0.v, false, bl0.v, (short)0, c00, false, false);
    c00 = __builtin_amdgcn_wmma_f32_16x16x32_bf16(false, aa0.v, false, br0.v, (short)0, c00, false, false);
    c01 = __builtin_amdgcn_wmma_f32_16x16x32_bf16(false, ax0.v, false, bl1.v, (short)0, c01, false, false);
    c01 = __builtin_amdgcn_wmma_f32_16x16x32_bf16(false, aa0.v, false, br1.v, (short)0, c01, false, false);
    c10 = __builtin_amdgcn_wmma_f32_16x16x32_bf16(false, ax1.v, false, bl0.v, (short)0, c10, false, false);
    c10 = __builtin_amdgcn_wmma_f32_16x16x32_bf16(false, aa1.v, false, br0.v, (short)0, c10, false, false);
    c11 = __builtin_amdgcn_wmma_f32_16x16x32_bf16(false, ax1.v, false, bl1.v, (short)0, c11, false, false);
    c11 = __builtin_amdgcn_wmma_f32_16x16x32_bf16(false, aa1.v, false, br1.v, (short)0, c11, false, false);
  }

  // ---------------- Store: C/D layout -> VGPR j = row hs*8+j, col = l ------
  const size_t ob = (size_t)rowbase * OUT_DIM;
#pragma unroll
  for (int j = 0; j < 8; ++j) {
    const int r0 = hs * 8 + j;        // row within first 16-row tile
    out[ob + (size_t)(r0)      * OUT_DIM + colb + l]      = c00[j];
    out[ob + (size_t)(r0)      * OUT_DIM + colb + 16 + l] = c01[j];
    out[ob + (size_t)(16 + r0) * OUT_DIM + colb + l]      = c10[j];
    out[ob + (size_t)(16 + r0) * OUT_DIM + colb + 16 + l] = c11[j];
  }
}

// ---------------------------------------------------------------------------
extern "C" void kernel_launch(void* const* d_in, const int* in_sizes, int n_in,
                              void* d_out, int out_size, void* d_ws, size_t ws_size,
                              hipStream_t stream) {
  const float* x     = (const float*)d_in[0];   // [T,N,IN]
  const float* neigh = (const float*)d_in[1];   // [T,N,K,IN]
  const float* wa    = (const float*)d_in[2];   // [1,K]
  const float* wl    = (const float*)d_in[3];   // [OUT,IN]
  const float* wr    = (const float*)d_in[4];   // [OUT,IN]
  float* out         = (float*)d_out;           // [M,OUT]

  bf16_t* wsb = (bf16_t*)d_ws;                  // 256 KB of bf16 weights

  pack_weights_bf16<<<(2 * OUT_DIM * IN_DIM) / 256, 256, 0, stream>>>(wl, wr, wsb);
  gat_fused_wmma<<<NUM_BLOCKS, 256, 0, stream>>>(
      x, neigh, wa, wsb, wsb + OUT_DIM * IN_DIM, out);
}
